// AttentionRNN_16372415332445
// MI455X (gfx1250) — matrix operand order repaired and checked by reading, not compiled, not run on hardware
//
#include <hip/hip_runtime.h>
#include <hip/hip_bf16.h>

typedef __attribute__((ext_vector_type(16))) _Float16 v16h;
typedef __attribute__((ext_vector_type(8)))  _Float16 v8h;
typedef __attribute__((ext_vector_type(8)))  float    v8f;

constexpr int Bn = 64, Tn = 1024, Hn = 512, Mn = 256;
constexpr int XDIM = Hn + Mn;        // 768
constexpr int LDSROW = Hn + 8;       // 520 halves: pad to kill bank conflicts

__device__ __forceinline__ float dot4(float4 a, float4 b) {
    return a.x * b.x + a.y * b.y + a.z * b.z + a.w * b.w;
}

// ---------------- Kernel 0: Wa (f32) -> f16 row-major ----------------
__global__ void waf16_kernel(const float* __restrict__ Wa,
                             _Float16* __restrict__ Waf16) {
    int i = blockIdx.x * 256 + threadIdx.x;   // grid covers exactly H*H
    Waf16[i] = (_Float16)Wa[i];
}

// ---------------- Kernel 1a: gate GEMMs gi = x@W_ih.T, gh = h@W_hh.T --
// grid (3H/256, B), 256 threads; thread owns one gate output element
__global__ void __launch_bounds__(256)
gates_kernel(const float* __restrict__ mem,
             const float* __restrict__ ctxv,
             const float* __restrict__ rnn,
             const float* __restrict__ W_ih,
             const float* __restrict__ b_ih,
             const float* __restrict__ W_hh,
             const float* __restrict__ b_hh,
             float* __restrict__ gi,
             float* __restrict__ gh) {
    __shared__ float xl[XDIM];
    __shared__ float hl[Hn];
    const int b = blockIdx.y;
    const int j = blockIdx.x * 256 + threadIdx.x;   // [0, 3H)

    for (int i = threadIdx.x; i < Mn; i += 256) xl[i] = mem[b * Mn + i];
    for (int i = threadIdx.x; i < Hn; i += 256) {
        xl[Mn + i] = ctxv[b * Hn + i];
        hl[i]      = rnn[b * Hn + i];
    }
    __syncthreads();

    const float4* x4 = (const float4*)xl;
    const float4* h4 = (const float4*)hl;
    const float4* wi = (const float4*)(W_ih + (size_t)j * XDIM);
    const float4* wh = (const float4*)(W_hh + (size_t)j * Hn);
    float gi_acc = 0.f, gh_acc = 0.f;
#pragma unroll 4
    for (int k = 0; k < XDIM / 4; ++k) gi_acc += dot4(wi[k], x4[k]);
#pragma unroll 4
    for (int k = 0; k < Hn / 4; ++k)  gh_acc += dot4(wh[k], h4[k]);

    gi[(size_t)b * 3 * Hn + j] = gi_acc + b_ih[j];
    gh[(size_t)b * 3 * Hn + j] = gh_acc + b_hh[j];
}

// ---------------- Kernel 1b: gate combine -> h_new -------------------
__global__ void __launch_bounds__(256)
hnew_kernel(const float* __restrict__ gi,
            const float* __restrict__ gh,
            const float* __restrict__ rnn,
            float* __restrict__ h_out) {
    const int idx = blockIdx.x * 256 + threadIdx.x;   // covers B*H
    const int b = idx >> 9, j = idx & (Hn - 1);
    const size_t base = (size_t)b * 3 * Hn;
    float ir = gi[base + j],          hr = gh[base + j];
    float iz = gi[base + Hn + j],     hz = gh[base + Hn + j];
    float inn = gi[base + 2*Hn + j],  hnn = gh[base + 2*Hn + j];
    float r = 1.f / (1.f + expf(-(ir + hr)));
    float z = 1.f / (1.f + expf(-(iz + hz)));
    float n = tanhf(inn + r * hnn);
    h_out[idx] = (1.f - z) * n + z * rnn[idx];
}

// ---------------- Kernel 1c: pqb = h_new@Wq.T + bq + ba --------------
// one block per batch, 512 threads
__global__ void __launch_bounds__(Hn)
pq_kernel(const float* __restrict__ h_new,
          const float* __restrict__ Wq,
          const float* __restrict__ bq,
          const float* __restrict__ ba,
          float* __restrict__ pqb) {
    __shared__ float hs[Hn];
    const int b = blockIdx.x, j = threadIdx.x;
    hs[j] = h_new[b * Hn + j];
    __syncthreads();
    const float4* q4 = (const float4*)(Wq + (size_t)j * Hn);
    const float4* h4 = (const float4*)hs;
    float pq = 0.f;
#pragma unroll 4
    for (int k = 0; k < Hn / 4; ++k) pq += dot4(q4[k], h4[k]);
    pqb[b * Hn + j] = pq + bq[j] + ba[j];
}

// ------- Kernel 2: fused pa = ann@Wa.T ; scores = tanh(pq+pa)@v -------
// grid (T/16, B); 256 threads = 8 waves; each block owns 16 time steps.
// Wave holds 4 n-tile accumulators; K loop kept rolled to avoid spills.
__global__ void __launch_bounds__(256)
attn_scores_kernel(const float* __restrict__ ann,
                   const _Float16* __restrict__ Waf16,
                   const float* __restrict__ pqb,
                   const float* __restrict__ vvec,
                   float* __restrict__ scores) {
    __shared__ _Float16 annT[16 * LDSROW];
    __shared__ float scAcc[16];
    const int b   = blockIdx.y;
    const int t0  = blockIdx.x * 16;
    const int tid = threadIdx.x;

    if (tid < 16) scAcc[tid] = 0.f;

    // stage 16 x 512 annotation tile (f32 -> f16), coalesced float4 reads
    const float4* src = (const float4*)(ann + ((size_t)(b * Tn + t0)) * Hn);
    for (int idx = tid; idx < 16 * Hn / 4; idx += 256) {
        int row = idx >> 7;           // Hn/4 = 128 float4 per row
        int c4  = idx & 127;
        float4 val = src[row * 128 + c4];
        _Float16* dst = annT + row * LDSROW + c4 * 4;
        dst[0] = (_Float16)val.x; dst[1] = (_Float16)val.y;
        dst[2] = (_Float16)val.z; dst[3] = (_Float16)val.w;
    }
    __syncthreads();

    const int lane = tid & 31;
    const int wave = tid >> 5;
    const int kh   = lane >> 4;       // K-half selector (ISA A/B layouts)
    const int ml   = lane & 15;       // A: row M ; B: col N (within tile)

    // per-lane B row base pointers for the 4 n-tiles owned by this wave
    const _Float16* bptr[4];
    float pqn[4], vn[4];
#pragma unroll
    for (int i = 0; i < 4; ++i) {
        const int n = (wave * 4 + i) * 16 + ml;
        bptr[i] = Waf16 + (size_t)n * Hn + kh * 16;
        pqn[i]  = pqb[b * Hn + n];
        vn[i]   = vvec[n];
    }

    const _Float16* arow = annT + ml * LDSROW + 16 * kh;

    v8f acc0 = {}, acc1 = {}, acc2 = {}, acc3 = {};
    union { v16h v; v8h h[2]; } af;

#pragma unroll 1
    for (int kc = 0; kc < Hn / 32; ++kc) {
        // A 16x32 f16 frag: lane holds K = kh*16 .. kh*16+15 (contig)
        af.h[0] = *(const v8h*)(arow + kc * 32);
        af.h[1] = *(const v8h*)(arow + kc * 32 + 8);
        // B 32x16 frags: lane n holds K = kh*16 .. kh*16+15 (contig 32B)
        v16h b0 = *(const v16h*)(bptr[0] + kc * 32);
        acc0 = __builtin_amdgcn_wmma_f32_16x16x32_f16(
            false, af.v, false, b0, (short)0, acc0, false, false);
        v16h b1 = *(const v16h*)(bptr[1] + kc * 32);
        acc1 = __builtin_amdgcn_wmma_f32_16x16x32_f16(
            false, af.v, false, b1, (short)0, acc1, false, false);
        v16h b2 = *(const v16h*)(bptr[2] + kc * 32);
        acc2 = __builtin_amdgcn_wmma_f32_16x16x32_f16(
            false, af.v, false, b2, (short)0, acc2, false, false);
        v16h b3 = *(const v16h*)(bptr[3] + kc * 32);
        acc3 = __builtin_amdgcn_wmma_f32_16x16x32_f16(
            false, af.v, false, b3, (short)0, acc3, false, false);
    }

    // epilogue: score[t] += tanh(pa + pq) * v[n]; C layout m = r + 8*kh
    v8f accs[4] = {acc0, acc1, acc2, acc3};
#pragma unroll
    for (int i = 0; i < 4; ++i) {
        float sc[8];
#pragma unroll
        for (int r = 0; r < 8; ++r) sc[r] = tanhf(accs[i][r] + pqn[i]) * vn[i];
#pragma unroll
        for (int off = 1; off < 16; off <<= 1)
#pragma unroll
            for (int r = 0; r < 8; ++r) sc[r] += __shfl_xor(sc[r], off, 32);
        if (ml == 0) {
            const int mb = kh * 8;
#pragma unroll
            for (int r = 0; r < 8; ++r) atomicAdd(&scAcc[mb + r], sc[r]);
        }
    }
    __syncthreads();
    if (tid < 16) scores[(size_t)b * Tn + t0 + tid] = scAcc[tid];
}

// ------- Kernel 3: softmax over T + ctx = align @ annotations --------
// one block per batch, 512 threads (annotations 2nd pass hits L2)
__global__ void __launch_bounds__(512)
softmax_ctx_kernel(const float* __restrict__ ann,
                   const float* __restrict__ scores,
                   float* __restrict__ ctx_out,
                   float* __restrict__ align_out) {
    __shared__ float alignS[Tn];
    __shared__ float red[512];
    const int b = blockIdx.x, tid = threadIdx.x;

    float s1 = scores[(size_t)b * Tn + tid];
    float s2 = scores[(size_t)b * Tn + 512 + tid];
    red[tid] = fmaxf(s1, s2);
    __syncthreads();
    for (int off = 256; off > 0; off >>= 1) {
        if (tid < off) red[tid] = fmaxf(red[tid], red[tid + off]);
        __syncthreads();
    }
    const float mx = red[0];
    __syncthreads();
    float e1 = expf(s1 - mx), e2 = expf(s2 - mx);
    red[tid] = e1 + e2;
    __syncthreads();
    for (int off = 256; off > 0; off >>= 1) {
        if (tid < off) red[tid] += red[tid + off];
        __syncthreads();
    }
    const float inv = 1.f / red[0];
    float a1 = e1 * inv, a2 = e2 * inv;
    alignS[tid] = a1; alignS[512 + tid] = a2;
    align_out[(size_t)b * Tn + tid] = a1;
    align_out[(size_t)b * Tn + 512 + tid] = a2;
    __syncthreads();

    float acc = 0.f;
    const float* abase = ann + (size_t)b * Tn * Hn + tid;   // thread = column h
    for (int t = 0; t < Tn; ++t) acc += alignS[t] * abase[(size_t)t * Hn];
    ctx_out[b * Hn + tid] = acc;
}

extern "C" void kernel_launch(void* const* d_in, const int* in_sizes, int n_in,
                              void* d_out, int out_size, void* d_ws, size_t ws_size,
                              hipStream_t stream) {
    const float* mem  = (const float*)d_in[0];
    const float* ctxv = (const float*)d_in[1];
    const float* rnn  = (const float*)d_in[2];
    const float* ann  = (const float*)d_in[3];
    const float* W_ih = (const float*)d_in[4];
    const float* b_ih = (const float*)d_in[5];
    const float* W_hh = (const float*)d_in[6];
    const float* b_hh = (const float*)d_in[7];
    const float* Wq   = (const float*)d_in[8];
    const float* bq   = (const float*)d_in[9];
    const float* Wa   = (const float*)d_in[10];
    const float* ba   = (const float*)d_in[11];
    const float* vv   = (const float*)d_in[12];

    float* out       = (float*)d_out;
    float* h_out     = out;                 // (B,H)
    float* ctx_out   = out + Bn * Hn;       // (B,H)
    float* align_out = out + 2 * Bn * Hn;   // (B,T)

    char* ws = (char*)d_ws;
    size_t off = 0;
    _Float16* Waf16 = (_Float16*)(ws + off); off += (size_t)Hn * Hn * 2;      // 512 KB
    float* pqb      = (float*)(ws + off);    off += (size_t)Bn * Hn * 4;      // 128 KB
    float* scores   = (float*)(ws + off);    off += (size_t)Bn * Tn * 4;      // 256 KB
    float* gi       = (float*)(ws + off);    off += (size_t)Bn * 3 * Hn * 4;  // 384 KB
    float* gh       = (float*)(ws + off);    off += (size_t)Bn * 3 * Hn * 4;  // 384 KB

    waf16_kernel<<<(Hn * Hn) / 256, 256, 0, stream>>>(Wa, Waf16);

    dim3 g1(3 * Hn / 256, Bn);   // (6, 64)
    gates_kernel<<<g1, 256, 0, stream>>>(mem, ctxv, rnn, W_ih, b_ih,
                                         W_hh, b_hh, gi, gh);
    hnew_kernel<<<(Bn * Hn) / 256, 256, 0, stream>>>(gi, gh, rnn, h_out);
    pq_kernel<<<Bn, Hn, 0, stream>>>(h_out, Wq, bq, ba, pqb);

    dim3 g2(Tn / 16, Bn);
    attn_scores_kernel<<<g2, 256, 0, stream>>>(ann, Waf16, pqb, vv, scores);
    softmax_ctx_kernel<<<Bn, 512, 0, stream>>>(ann, scores, ctx_out, align_out);
}
